// AdaConv2D_87179246174619
// MI455X (gfx1250) — matrix operand order repaired and checked by reading, not compiled, not run on hardware
//
#include <hip/hip_runtime.h>
#include <hip/hip_bf16.h>

#define BATCH 8
#define CH    512
#define HH    128
#define WW    128
#define EPSV  1e-5f
#define SLOPE 0.01f

typedef __attribute__((ext_vector_type(16))) __bf16 v16bf;
typedef __attribute__((ext_vector_type(8)))  float  v8f;
typedef __attribute__((ext_vector_type(4)))  unsigned int u32x4;
typedef __attribute__((ext_vector_type(8)))  int    i32x8;
typedef __attribute__((ext_vector_type(4)))  int    i32x4;

union FragB { uint4 u[2]; v16bf v; };

// workspace layout (bytes)
static constexpr size_t Y_OFF  = 0;                                   // bf16 [B][H][W][C]
static constexpr size_t WT_OFF = (size_t)BATCH * HH * WW * CH * 2;    // bf16 [9][CO][CI]
static constexpr size_t ST_OFF = WT_OFF + (size_t)9 * CH * CH * 2;    // f32  [B][C][2] mean,rstd

// ---------------------------------------------------------------------------
// Kernel 0: pack spatial_w [co][ci][3][3] f32 -> wt[tap][co][ci] bf16
// ---------------------------------------------------------------------------
__global__ void wpack_kernel(const float* __restrict__ sw,
                             __hip_bfloat16* __restrict__ wt) {
  int idx = blockIdx.x * 256 + threadIdx.x;
  if (idx >= 9 * CH * CH) return;
  int tap = idx >> 18;             // / (512*512)
  int rem = idx & (CH * CH - 1);
  int co  = rem >> 9;
  int ci  = rem & (CH - 1);
  wt[idx] = __float2bfloat16(sw[((size_t)co * CH + ci) * 9 + tap]);
}

// ---------------------------------------------------------------------------
// Kernel 1: per-(b,c) instance-norm statistics (wave32 shfl reduction)
// ---------------------------------------------------------------------------
__global__ void stats_kernel(const float* __restrict__ x, float* __restrict__ st) {
  const int c = blockIdx.x, b = blockIdx.y;
  const float4* p = (const float4*)(x + (size_t)(b * CH + c) * (HH * WW));
  float s1 = 0.f, s2 = 0.f;
  for (int i = threadIdx.x; i < HH * WW / 4; i += 256) {
    float4 v = p[i];
    s1 += v.x + v.y + v.z + v.w;
    s2 += v.x * v.x + v.y * v.y + v.z * v.z + v.w * v.w;
  }
  #pragma unroll
  for (int off = 16; off; off >>= 1) {
    s1 += __shfl_xor(s1, off, 32);
    s2 += __shfl_xor(s2, off, 32);
  }
  __shared__ float r1[8], r2[8];
  int lane = threadIdx.x & 31, wv = threadIdx.x >> 5;
  if (lane == 0) { r1[wv] = s1; r2[wv] = s2; }
  __syncthreads();
  if (threadIdx.x == 0) {
    float t1 = 0.f, t2 = 0.f;
    #pragma unroll
    for (int i = 0; i < 8; ++i) { t1 += r1[i]; t2 += r2[i]; }
    const float inv = 1.0f / (HH * WW);
    float mean = t1 * inv;
    float var  = t2 * inv - mean * mean;
    st[(b * CH + c) * 2 + 0] = mean;
    st[(b * CH + c) * 2 + 1] = rsqrtf(var + EPSV);
  }
}

// ---------------------------------------------------------------------------
// Kernel 2: fused normalize + dynamic depthwise 3x3 (reflect) + pw*scale+bias
//           writes bf16 NHWC intermediate y[b][h][w][c]
// conv(s*(x-mean)) = s*conv(x) - s*mean*sum(k)
// ---------------------------------------------------------------------------
__global__ void dwsep_kernel(const float* __restrict__ x,
                             const float* __restrict__ dw,
                             const float* __restrict__ pw,
                             const float* __restrict__ bias,
                             const float* __restrict__ st,
                             __hip_bfloat16* __restrict__ y) {
  const int chunk = blockIdx.x;          // 16 chunks of 8 rows
  const int c = blockIdx.y, b = blockIdx.z;
  const int r0 = chunk * 8;
  const size_t plane = (size_t)(b * CH + c) * (HH * WW);

  __shared__ float tile[10 * 132];       // 10 rows x 130 cols (padded)
  for (int idx = threadIdx.x; idx < 10 * 130; idx += 256) {
    int lr = idx / 130, cc = idx % 130;
    int gr = r0 - 1 + lr;  gr = gr < 0 ? 1 : (gr > HH - 1 ? HH - 2 : gr);
    int gc = cc - 1;       gc = gc < 0 ? 1 : (gc > WW - 1 ? WW - 2 : gc);
    tile[lr * 132 + cc] = x[plane + gr * WW + gc];
  }
  __syncthreads();

  const int bc = b * CH + c;             // uniform -> scalar loads
  float k[9], ks = 0.f;
  #pragma unroll
  for (int i = 0; i < 9; ++i) { k[i] = dw[bc * 9 + i]; ks += k[i]; }
  const float mean  = st[bc * 2 + 0];
  const float rstd  = st[bc * 2 + 1];
  const float scale = pw[bc] * rstd;
  const float shift = bias[bc] - scale * mean * ks;

  #pragma unroll
  for (int t = 0; t < 4; ++t) {
    int px = threadIdx.x + t * 256;      // 8*128 pixels / 256 threads
    int r = px >> 7, w = px & 127;
    const float* trow = tile + r * 132 + w;
    float acc = trow[0]   * k[0] + trow[1]   * k[1] + trow[2]   * k[2]
              + trow[132] * k[3] + trow[133] * k[4] + trow[134] * k[5]
              + trow[264] * k[6] + trow[265] * k[7] + trow[266] * k[8];
    float val = scale * acc + shift;
    y[(((size_t)(b * HH + r0 + r)) * WW + w) * CH + c] = __float2bfloat16(val);
  }
}

// ---------------------------------------------------------------------------
// TDM descriptor: 2D tile (tile_dim0 = 64 ci elems, tile_dim1 = 256 co rows)
// from wt[tap][co][ci] (stride 512 elems, 2-byte data) into LDS (row = 128B).
// Bitfields per CDNA5 ISA section 8.3/8.4. 6-arg builtin (clang-23 lane).
// ---------------------------------------------------------------------------
__device__ static inline void tdm_load_b(unsigned lds_off, unsigned long long ga) {
  u32x4 g0;
  g0[0] = 1u;                                                 // count=1, user desc
  g0[1] = lds_off;                                            // lds_addr [63:32]
  g0[2] = (unsigned)(ga & 0xffffffffu);                       // global_addr[31:0]
  g0[3] = (unsigned)((ga >> 32) & 0x01ffffffu) | (2u << 30);  // ga[56:32] | type=2
  i32x8 g1;
  g1[0] = (int)(1u << 16);          // workgroup_mask=0, data_size=1 (2 bytes)
  g1[1] = (int)(512u << 16);        // tensor_dim0 = 512  (bits 79:48)
  g1[2] = (int)(512u << 16);        // tensor_dim1 = 512  (bits 111:80)
  g1[3] = (int)(64u << 16);         // tile_dim0 = 64     (bits 127:112)
  g1[4] = 256;                      // tile_dim1 = 256, tile_dim2 = 0
  g1[5] = 512;                      // tensor_dim0_stride = 512 (bits 207:160)
  g1[6] = 0;
  g1[7] = 0;                        // tensor_dim1_stride = 0 (unused, 2D)
  i32x4 z4 = {0, 0, 0, 0};
  i32x8 z8 = {0, 0, 0, 0, 0, 0, 0, 0};
  __builtin_amdgcn_tensor_load_to_lds(g0, g1, z4, z4, z8, 0);
}

// ---------------------------------------------------------------------------
// Kernel 3: dense 3x3 conv (reflect) as implicit GEMM on v_wmma_f32_16x16x32_bf16.
// Workgroup: 8 waves = 32 px x 256 co; wave = 16 px x 64 co (4 accumulators).
// Weights: TDM tensor_load_to_lds, double-buffered (2 x 32 KB), TENSORcnt sync.
// Activations: direct b128 global loads (NHWC, fragment-shaped).
// ---------------------------------------------------------------------------
#define NIT (9 * 8)                 // 9 taps x 8 k-blocks of 64

__global__ void __launch_bounds__(256) conv_wmma_kernel(
    const __hip_bfloat16* __restrict__ yb,
    const __hip_bfloat16* __restrict__ wtb,
    const float* __restrict__ sb,
    float* __restrict__ out) {
  __shared__ char bbuf[2][32768];   // 256 co rows x 128 B (64 ci bf16), x2

  const int lane = threadIdx.x & 31;
  const int wave = threadIdx.x >> 5;
  const int pg = wave >> 2;                 // pixel group 0..1
  const int cg = wave & 3;                  // channel group 0..3
  const int t  = blockIdx.x;                // (b, h, w-tile of 32)
  const int wt32 = t & 3;
  const int h    = (t >> 2) & (HH - 1);
  const int b    = t >> 9;
  const int w0   = wt32 * 32 + pg * 16;
  const int co0  = blockIdx.y * 256;
  const int m  = lane & 15;                 // A row M / B col N
  const int hi = lane >> 4;
  const int wpix = w0 + m;

  const unsigned long long wbase =
      (unsigned long long)(uintptr_t)wtb + ((unsigned long long)co0 << 10);

  if (wave == 0)                            // prologue: fill buffer 0 (tap 0, k0 0)
    tdm_load_b((unsigned)(unsigned long long)(uintptr_t)&bbuf[0][0], wbase);

  v8f a0 = {}, a1 = {}, a2 = {}, a3 = {};

  for (int it = 0; it < NIT; ++it) {
    const int tap = (unsigned)it >> 3;
    const int k0  = (it & 7) << 6;

    if (wave == 0) __builtin_amdgcn_s_wait_tensorcnt(0);
    __syncthreads();                        // bbuf[it&1] ready for everyone

    if (wave == 0 && it + 1 < NIT) {        // prefetch next tile via TDM
      const int nx = it + 1;
      const unsigned long long ga = wbase
          + ((unsigned long long)((unsigned)nx >> 3) << 19)   // tap * 512*512*2
          + (unsigned long long)(((nx & 7) << 6) << 1);       // k0 * 2
      tdm_load_b((unsigned)(unsigned long long)(uintptr_t)&bbuf[(it + 1) & 1][0], ga);
    }

    const int kh = tap / 3 - 1, kw = tap % 3 - 1;
    int hh = h + kh;    hh = hh < 0 ? -hh : (hh > HH - 1 ? 2 * HH - 2 - hh : hh);
    int ww = wpix + kw; ww = ww < 0 ? -ww : (ww > WW - 1 ? 2 * WW - 2 - ww : ww);
    const __hip_bfloat16* yrow = yb + (((size_t)(b * HH + hh)) * WW + ww) * CH + k0;
    const char* bb = bbuf[it & 1] + (cg * 64 + m) * 128 + 32 * hi;

    #pragma unroll
    for (int s = 0; s < 2; ++s) {           // two K=32 steps per 64-ci tile
      FragB af;
      af.u[0] = *(const uint4*)(yrow + s * 32 + 16 * hi);
      af.u[1] = *(const uint4*)(yrow + s * 32 + 16 * hi + 8);

      const char* bs = bb + s * 64;         // B fragments from LDS (ds_load_b128)
      FragB f0, f1, f2, f3;
      f0.u[0] = *(const uint4*)(bs);            f0.u[1] = *(const uint4*)(bs + 16);
      f1.u[0] = *(const uint4*)(bs + 2048);     f1.u[1] = *(const uint4*)(bs + 2064);
      f2.u[0] = *(const uint4*)(bs + 4096);     f2.u[1] = *(const uint4*)(bs + 4112);
      f3.u[0] = *(const uint4*)(bs + 6144);     f3.u[1] = *(const uint4*)(bs + 6160);

      a0 = __builtin_amdgcn_wmma_f32_16x16x32_bf16(false, af.v, false, f0.v, (short)0, a0, false, false);
      a1 = __builtin_amdgcn_wmma_f32_16x16x32_bf16(false, af.v, false, f1.v, (short)0, a1, false, false);
      a2 = __builtin_amdgcn_wmma_f32_16x16x32_bf16(false, af.v, false, f2.v, (short)0, a2, false, false);
      a3 = __builtin_amdgcn_wmma_f32_16x16x32_bf16(false, af.v, false, f3.v, (short)0, a3, false, false);
    }
  }

  // Epilogue: D layout -> lane N=m owns co, vgpr v -> pixel w0 + v + 8*hi.
  v8f accs[4] = {a0, a1, a2, a3};
  #pragma unroll
  for (int nt = 0; nt < 4; ++nt) {
    const int co = co0 + cg * 64 + nt * 16 + m;
    const float bv = sb[co];
    float e[8];
    #pragma unroll
    for (int i = 0; i < 8; ++i) {
      float v = accs[nt][i] + bv;
      e[i] = v >= 0.f ? v : SLOPE * v;
    }
    float* obase = out + (((size_t)(b * CH + co)) * HH + h) * WW + w0 + 8 * hi;
    *(float4*)(obase)     = make_float4(e[0], e[1], e[2], e[3]);
    *(float4*)(obase + 4) = make_float4(e[4], e[5], e[6], e[7]);
  }
}

// ---------------------------------------------------------------------------
extern "C" void kernel_launch(void* const* d_in, const int* in_sizes, int n_in,
                              void* d_out, int out_size, void* d_ws, size_t ws_size,
                              hipStream_t stream) {
  const float* x    = (const float*)d_in[0];   // [8,512,128,128]
  const float* dw   = (const float*)d_in[1];   // [8,512,1,3,3]
  const float* pw   = (const float*)d_in[2];   // [8,512,1,1,1]
  const float* bias = (const float*)d_in[3];   // [8,512]
  const float* sw   = (const float*)d_in[4];   // [512,512,3,3]
  const float* sb   = (const float*)d_in[5];   // [512]
  float* out = (float*)d_out;

  char* ws = (char*)d_ws;
  __hip_bfloat16* y  = (__hip_bfloat16*)(ws + Y_OFF);
  __hip_bfloat16* wt = (__hip_bfloat16*)(ws + WT_OFF);
  float*          st = (float*)(ws + ST_OFF);

  wpack_kernel<<<(9 * CH * CH) / 256, 256, 0, stream>>>(sw, wt);
  stats_kernel<<<dim3(CH, BATCH), 256, 0, stream>>>(x, st);
  dwsep_kernel<<<dim3(16, CH, BATCH), 256, 0, stream>>>(x, dw, pw, bias, st, y);
  conv_wmma_kernel<<<dim3(4096, 2), 256, 0, stream>>>(y, wt, sb, out);
}
